// Reverb_79834852098096
// MI455X (gfx1250) — matrix operand order repaired and checked by reading, not compiled, not run on hardware
//
#include <hip/hip_runtime.h>

// ---------------------------------------------------------------------------
// Reverb = 44100-tap FIR as banded-Toeplitz WMMA GEMM (gfx1250, wave32).
//
//   out[t0+16m+n] = sum_j sum_r A_j[m,r] * B_j[r,n]
//   A_j[m,r] = xpad[t0 + 16m + 32j + r]          (padded f16 signal)
//   B_j[r,n] = h[32j + r - n]  (0 outside [1,K))  (pre-swizzled f16 table)
//
// Dual-tile waves: each wave owns tiles t0 and t0+256. Since
// A_j(t0+256) == A_{j+8}(t0), every A block loaded from global feeds TWO
// v_wmma_f32_16x16x32_f16 ops (acc0 with the current LDS filter chunk, acc1
// with the previous, double-buffered chunk) -> 1 global_load_b128 per WMMA.
// ---------------------------------------------------------------------------

typedef __attribute__((ext_vector_type(16))) _Float16 v16h;
typedef __attribute__((ext_vector_type(8)))  _Float16 v8h;
typedef __attribute__((ext_vector_type(8)))  float    v8f;

#define SIG_LEN   220500          // samples per batch row
#define BATCH     4
#define IR_LEN    44100           // taps (tap 0 forced to zero by reference)
#define LEFT      22049           // (IR_LEN-1)/2 left pad
#define TP        264960          // padded signal length per batch
#define NJ        1384            // 32-tap blocks per tile (multiple of 8)
#define NJ_ALLOC  1392            // +8 zero blocks for acc1's lagged chunk
#define JCHUNK    8               // filter blocks staged in LDS per pass
#define PAIRS_PB  431             // tile-pairs per batch (431*2*256 >= 220500)
#define WG_PB     54              // ceil(431/8) workgroups per batch

// ---------------------------------------------------------------------------
// Prep 1: zero-padded f16 copy of the signal. xp[b][i] = x[b][i - LEFT].
// ---------------------------------------------------------------------------
__global__ void prep_pad(const float* __restrict__ sig, _Float16* __restrict__ xp) {
  int idx = blockIdx.x * blockDim.x + threadIdx.x;
  if (idx >= BATCH * TP) return;
  int b = idx / TP;
  int i = idx - b * TP;
  int s = i - LEFT;
  float v = (s >= 0 && s < SIG_LEN) ? sig[(size_t)b * SIG_LEN + s] : 0.0f;
  xp[idx] = (_Float16)v;
}

// ---------------------------------------------------------------------------
// Prep 2: Toeplitz filter blocks in per-lane WMMA B layout.
// btab[j][lane][hi]: lane holds column n = lane&15; local K = (hi&7)+16*(hi>>3)+8*(lane>>4);
// k = 32j + K - n.  k outside [1, IR_LEN) (incl. zeroed dry tap k==0) -> 0.
// Blocks j >= 1379 are naturally all-zero (k >= IR_LEN).
// ---------------------------------------------------------------------------
__global__ void prep_btab(const float* __restrict__ ir, _Float16* __restrict__ btab) {
  int idx = blockIdx.x * blockDim.x + threadIdx.x;
  if (idx >= NJ_ALLOC * 512) return;
  int j    = idx >> 9;
  int s    = idx & 511;
  int lane = s >> 4;
  int hi   = s & 15;
  int n    = lane & 15;
  int kloc = (hi & 7) + ((hi >> 3) << 4) + ((lane >> 4) << 3);
  int k    = 32 * j + kloc - n;
  float v = (k >= 1 && k < IR_LEN) ? ir[k] : 0.0f;
  btab[idx] = (_Float16)v;
}

// ---------------------------------------------------------------------------
// Main: 8 waves/WG, each wave = two adjacent 16x16 output tiles.
// ---------------------------------------------------------------------------
__global__ __launch_bounds__(256) void reverb_wmma(
    const _Float16* __restrict__ xp,    // [BATCH][TP]
    const _Float16* __restrict__ btab,  // [NJ_ALLOC][32][16] halves
    float* __restrict__ out)            // [BATCH][SIG_LEN]
{
  __shared__ __align__(32) _Float16 bsh[2][JCHUNK][512];  // 16 KB double buffer

  const int tid   = threadIdx.x;
  const int lane  = tid & 31;
  const int wave  = tid >> 5;
  const int batch = blockIdx.x / WG_PB;
  const int wg    = blockIdx.x % WG_PB;
  const int pair  = wg * 8 + wave;

  int t0 = pair * 512;                 // acc0 tile base; acc1 = t0 + 256
  const bool active = (t0 < SIG_LEN);
  if (!active) t0 = 0;                 // dead tail waves: compute garbage,
                                       // store nothing (EXEC stays all-1s)

  const _Float16* __restrict__ xpb = xp + (size_t)batch * TP;
  // A per-lane base: row m = lane&15; K-stripe offset 8*(lane>>4).
  // Lane's 16 halves = two contiguous 8-half runs at abase, abase+16.
  const int abase = t0 + 16 * (lane & 15) + 8 * (lane >> 4);

  v8f acc0 = {};
  v8f acc1 = {};

  // zero both LDS buffers so chunk 0's "previous" chunk contributes 0 to acc1
  {
    uint4 zz = {0u, 0u, 0u, 0u};
    uint4* z = (uint4*)&bsh[0][0][0];
#pragma unroll
    for (int k = 0; k < 4; ++k) z[tid + k * 256] = zz;   // 1024 uint4 = 16 KB
  }

  const uint4* __restrict__ gsrc = (const uint4*)btab;   // 64 uint4 / j-block
  int c = 0;
  for (int jc = 0; jc <= NJ; jc += JCHUNK, ++c) {        // 174 chunks
    const int buf = c & 1;
    __syncthreads();
    // cooperative stage of 8 filter blocks (8 KB): 256 threads x 32 B
    const uint4* src = gsrc + (size_t)jc * 64;
    uint4* ldst = (uint4*)&bsh[buf][0][0];
    ldst[tid]       = src[tid];
    ldst[tid + 256] = src[tid + 256];
    __builtin_prefetch((const void*)(src + 512), 0, 1);  // next chunk -> L2
    __syncthreads();

#pragma unroll
    for (int jj = 0; jj < JCHUNK; ++jj) {
      const int s0 = abase + 32 * (jc + jj);
      union { v16h v; v8h h[2]; } a;
      a.h[0] = *(const v8h*)(xpb + s0);        // K 0..7   (+stripe)
      a.h[1] = *(const v8h*)(xpb + s0 + 16);   // K 16..23 (+stripe)
      // acc0: tile t0 with block j = jc+jj (current chunk)
      v16h bc = *(const v16h*)&bsh[buf][jj][lane << 4];
      acc0 = __builtin_amdgcn_wmma_f32_16x16x32_f16(
          false, a.v, false, bc, (short)0, acc0, false, false);
      // acc1: tile t0+256 with block j-8 (previous chunk, same A!)
      v16h bp = *(const v16h*)&bsh[buf ^ 1][jj][lane << 4];
      acc1 = __builtin_amdgcn_wmma_f32_16x16x32_f16(
          false, a.v, false, bp, (short)0, acc1, false, false);
    }
  }

  if (active) {
    // C/D layout: VGPR g, lane -> row M = g + 8*(lane>>4), col N = lane&15
    float* __restrict__ ob = out + (size_t)batch * SIG_LEN;
    const int rb0 = t0 + ((lane >> 4) << 7) + (lane & 15);
#pragma unroll
    for (int g = 0; g < 8; ++g) {
      int t = rb0 + (g << 4);
      if (t < SIG_LEN) ob[t] = acc0[g];
    }
    const int rb1 = rb0 + 256;
#pragma unroll
    for (int g = 0; g < 8; ++g) {
      int t = rb1 + (g << 4);
      if (t < SIG_LEN) ob[t] = acc1[g];
    }
  }
}

// ---------------------------------------------------------------------------
// Launch
// ---------------------------------------------------------------------------
extern "C" void kernel_launch(void* const* d_in, const int* in_sizes, int n_in,
                              void* d_out, int out_size, void* d_ws, size_t ws_size,
                              hipStream_t stream) {
  const float* sig = (const float*)d_in[0];   // [4][220500] f32
  const float* ir  = (const float*)d_in[1];   // [44100] f32
  float* out = (float*)d_out;                 // [4][220500] f32

  // workspace: [ btab: NJ_ALLOC*512 halves | xp: BATCH*TP halves ]  (~3.5 MB)
  _Float16* btab = (_Float16*)d_ws;
  _Float16* xp   = (_Float16*)d_ws + (size_t)NJ_ALLOC * 512;

  prep_btab<<<(NJ_ALLOC * 512 + 255) / 256, 256, 0, stream>>>(ir, btab);
  prep_pad<<<(BATCH * TP + 255) / 256, 256, 0, stream>>>(sig, xp);
  reverb_wmma<<<BATCH * WG_PB, 256, 0, stream>>>(xp, btab, out);
}
